// LSTM_79980880986313
// MI455X (gfx1250) — compile-verified, matrix operand order checked by reading, not run on hardware
//
#include <hip/hip_runtime.h>

#define DEV __device__ __forceinline__

typedef __attribute__((ext_vector_type(16))) __bf16        v16bf;
typedef __attribute__((ext_vector_type(8)))  float         v8f;
typedef __attribute__((ext_vector_type(8)))  unsigned int  v8u;

// ---------------- problem constants ----------------
constexpr int B_   = 64;
constexpr int T_   = 512;
constexpr int I_   = 512;
constexpr int H_   = 1024;
constexpr int G_   = 4 * H_;      // 4096 gate columns
constexpr int K_   = I_ + H_;     // 1536 fused K (x | h)
constexpr int KB_X = I_ / 32;     // 16 WMMA K-blocks from x
constexpr int KB_H = H_ / 32;     // 32 WMMA K-blocks from h
constexpr int KB_  = K_ / 32;     // 48 total
constexpr int NT_  = G_ / 16;     // 256 N tiles
constexpr int MT_  = B_ / 16;     // 4 M tiles
constexpr int GRID = 128;         // persistent workgroups (x 8 waves = 1024 tiles)

// ---------------- workspace layout (bytes) ----------------
constexpr size_t OFF_WPACK = 0;
constexpr size_t SZ_WPACK  = (size_t)NT_ * KB_ * 32 * 8 * 4;          // 12.58 MB
constexpr size_t OFF_XPACK = OFF_WPACK + SZ_WPACK;
constexpr size_t SZ_XPACK  = (size_t)T_ * MT_ * KB_X * 32 * 8 * 4;    // 32 MB
constexpr size_t OFF_HPACK = OFF_XPACK + SZ_XPACK;
constexpr size_t SZ_HPACK  = (size_t)MT_ * KB_H * 32 * 8 * 4;         // 128 KB
constexpr size_t OFF_GATES = OFF_HPACK + SZ_HPACK;
constexpr size_t SZ_GATES  = (size_t)B_ * G_ * 4;                     // 1 MB
constexpr size_t OFF_C     = OFF_GATES + SZ_GATES;
constexpr size_t SZ_C      = (size_t)B_ * H_ * 4;                     // 256 KB
constexpr size_t OFF_BIAS  = OFF_C + SZ_C;
constexpr size_t SZ_BIAS   = (size_t)G_ * 4;                          // 16 KB
constexpr size_t OFF_BAR   = OFF_BIAS + SZ_BIAS;                      // 256 B

// ---------------- helpers ----------------
DEV unsigned short f2bf(float f) {            // fp32 -> bf16 round-to-nearest-even
    unsigned u = __builtin_bit_cast(unsigned, f);
    unsigned r = 0x7FFFu + ((u >> 16) & 1u);
    return (unsigned short)((u + r) >> 16);
}
DEV unsigned packbf(float lo, float hi) {
    return (unsigned)f2bf(lo) | ((unsigned)f2bf(hi) << 16);
}
// 16-bit WMMA operand fragment slot (CDNA5 ISA 7.12.2):
// within a 32-wide K block, K-pair q=(k&31)>>1 and row/col r=(m|n)&15 live at
//   v    = (q&3) | ((q>>3)<<2)       (dword index 0..7)
//   lane = ((q>>2)&1)*16 + r
DEV void fragslot(int q, int r, int& v, int& lane) {
    v    = (q & 3) | (((q >> 3) & 1) << 2);
    lane = (((q >> 2) & 1) << 4) | r;
}
DEV float sigf(float x) { return 1.0f / (1.0f + __expf(-x)); }

DEV void gbar(unsigned* cnt, unsigned target) {
    __syncthreads();
    if (threadIdx.x == 0) {
        __threadfence();                                   // release
        __hip_atomic_fetch_add(cnt, 1u, __ATOMIC_RELEASE, __HIP_MEMORY_SCOPE_AGENT);
        while (__hip_atomic_load(cnt, __ATOMIC_ACQUIRE, __HIP_MEMORY_SCOPE_AGENT) < target)
            __builtin_amdgcn_s_sleep(1);
    }
    __syncthreads();
    __threadfence();                                       // acquire (invalidate WGP$)
}

// ---------------- prep kernels ----------------
// Fused weight pack: Wc[k][n] = k<I ? W_ih[n][k] : W_hh[n][k-I]  -> B-fragment layout
__global__ void pack_w_kernel(const float* __restrict__ Wih,
                              const float* __restrict__ Whh,
                              unsigned* __restrict__ Wpack) {
    int tid = blockIdx.x * blockDim.x + threadIdx.x;
    if (tid >= G_ * (K_ / 2)) return;
    int n  = tid / (K_ / 2);
    int p  = tid % (K_ / 2);
    int k0 = p * 2, k1 = k0 + 1;
    float w0 = (k0 < I_) ? Wih[(size_t)n * I_ + k0] : Whh[(size_t)n * H_ + (k0 - I_)];
    float w1 = (k1 < I_) ? Wih[(size_t)n * I_ + k1] : Whh[(size_t)n * H_ + (k1 - I_)];
    int kb = k0 >> 5, q = (k0 & 31) >> 1;
    int v, lane; fragslot(q, n & 15, v, lane);
    int ntile = n >> 4;
    Wpack[(((size_t)ntile * KB_ + kb) * 32 + lane) * 8 + v] = packbf(w0, w1);
}

// Pre-pack the entire input sequence into A-fragment layout (read once, 32 MB bf16)
__global__ void pack_x_kernel(const float* __restrict__ seq,
                              unsigned* __restrict__ Xpack) {
    int tid = blockIdx.x * blockDim.x + threadIdx.x;
    if (tid >= T_ * B_ * (I_ / 2)) return;
    int t  = tid / (B_ * (I_ / 2));
    int r  = tid % (B_ * (I_ / 2));
    int b  = r / (I_ / 2);
    int p  = r % (I_ / 2);
    int k0 = p * 2;
    const float* x = seq + ((size_t)b * T_ + t) * I_;   // sequence[b][t][:]
    int kb = k0 >> 5, q = (k0 & 31) >> 1;
    int v, lane; fragslot(q, b & 15, v, lane);
    int mtile = b >> 4;
    Xpack[((((size_t)t * MT_ + mtile) * KB_X + kb) * 32 + lane) * 8 + v] =
        packbf(x[k0], x[k0 + 1]);
}

__global__ void bias_kernel(const float* __restrict__ bi,
                            const float* __restrict__ bh,
                            float* __restrict__ bias) {
    int i = blockIdx.x * blockDim.x + threadIdx.x;
    if (i < G_) bias[i] = bi[i] + bh[i];
}

// ---------------- persistent LSTM scan ----------------
__launch_bounds__(256, 1)
__global__ void lstm_scan_kernel(const unsigned* __restrict__ Wpack,
                                 const unsigned* __restrict__ Xpack,
                                 unsigned* __restrict__ Hpack,
                                 float* __restrict__ gates,
                                 float* __restrict__ cbuf,
                                 const float* __restrict__ bias,
                                 const int* __restrict__ lengths,
                                 float* __restrict__ out,
                                 unsigned* __restrict__ barcnt) {
    const int lane  = threadIdx.x & 31;
    const int wave  = threadIdx.x >> 5;
    const int w     = blockIdx.x * 8 + wave;   // 0..1023, one 16x16 gate tile each
    const int ntile = w >> 2;                  // 0..255
    const int mtile = w & 3;                   // 0..3

    // pointwise assignment: each thread owns a (batch row, even-k pair)
    const int tid = blockIdx.x * 256 + threadIdx.x;   // 0..32767
    const int pb  = tid >> 9;                         // 0..63
    const int pk0 = (tid & 511) * 2;                  // 0..1022 even
    int hv, hlane; fragslot((pk0 & 31) >> 1, pb & 15, hv, hlane);
    unsigned* hdst =
        Hpack + (((size_t)(pb >> 4) * KB_H + (pk0 >> 5)) * 32 + hlane) * 8 + hv;
    const int mylen = lengths[pb];

    const unsigned* wp = Wpack + ((size_t)ntile * KB_ * 32 + lane) * 8;
    const unsigned* hp = Hpack + ((size_t)mtile * KB_H * 32 + lane) * 8;

    unsigned round = 0;
    for (int t = 0; t < T_; ++t) {
        // ---- GEMM: gates[64,4096] = [x_t | h] x Wc, bf16 WMMA, f32 accum ----
        const unsigned* xp = Xpack + (((size_t)t * MT_ + mtile) * KB_X * 32 + lane) * 8;
        v8f acc0 = {}; v8f acc1 = {};
        #pragma unroll 4
        for (int kb = 0; kb < KB_X; ++kb) {
            v16bf Af = __builtin_bit_cast(v16bf, *(const v8u*)(xp + (size_t)kb * 256));
            v16bf Bf = __builtin_bit_cast(v16bf, *(const v8u*)(wp + (size_t)kb * 256));
            if (kb & 1)
                acc1 = __builtin_amdgcn_wmma_f32_16x16x32_bf16(false, Af, false, Bf,
                                                               (short)0, acc1, false, false);
            else
                acc0 = __builtin_amdgcn_wmma_f32_16x16x32_bf16(false, Af, false, Bf,
                                                               (short)0, acc0, false, false);
        }
        #pragma unroll 4
        for (int kb = 0; kb < KB_H; ++kb) {
            v16bf Af = __builtin_bit_cast(v16bf, *(const v8u*)(hp + (size_t)kb * 256));
            v16bf Bf = __builtin_bit_cast(v16bf,
                         *(const v8u*)(wp + (size_t)(KB_X + kb) * 256));
            if (kb & 1)
                acc1 = __builtin_amdgcn_wmma_f32_16x16x32_bf16(false, Af, false, Bf,
                                                               (short)0, acc1, false, false);
            else
                acc0 = __builtin_amdgcn_wmma_f32_16x16x32_bf16(false, Af, false, Bf,
                                                               (short)0, acc0, false, false);
        }
        v8f acc = acc0 + acc1;

        // C/D layout: dword rr -> M = rr + 8*(lane>=16), N = lane&15
        {
            const int nl = lane & 15;
            const int mh = (lane >> 4) * 8;
            float* gdst = gates + (size_t)(mtile * 16 + mh) * G_ + ntile * 16 + nl;
            #pragma unroll
            for (int rr = 0; rr < 8; ++rr) gdst[(size_t)rr * G_] = acc[rr];
        }
        gbar(barcnt, ++round * (unsigned)GRID);

        // ---- pointwise cell update; write h back in A-fragment layout ----
        {
            const float* gr = gates + (size_t)pb * G_;
            float h2[2];
            #pragma unroll
            for (int e = 0; e < 2; ++e) {
                int k = pk0 + e;
                float iv = sigf (gr[k]          + bias[k]);
                float fv = sigf (gr[H_ + k]     + bias[H_ + k]);
                float gv = tanhf(gr[2 * H_ + k] + bias[2 * H_ + k]);
                float ov = sigf (gr[3 * H_ + k] + bias[3 * H_ + k]);
                float cv = fv * cbuf[(size_t)pb * H_ + k] + iv * gv;
                cbuf[(size_t)pb * H_ + k] = cv;
                float hnew = ov * tanhf(cv);
                h2[e] = hnew;
                if (mylen > t) out[(size_t)pb * H_ + k] = hnew;
            }
            *hdst = packbf(h2[0], h2[1]);
        }
        gbar(barcnt, ++round * (unsigned)GRID);
    }
}

// ---------------- host launcher ----------------
extern "C" void kernel_launch(void* const* d_in, const int* in_sizes, int n_in,
                              void* d_out, int out_size, void* d_ws, size_t ws_size,
                              hipStream_t stream) {
    const float* seq = (const float*)d_in[0];
    const int*   len = (const int*)  d_in[1];
    const float* Wih = (const float*)d_in[2];
    const float* Whh = (const float*)d_in[3];
    const float* bih = (const float*)d_in[4];
    const float* bhh = (const float*)d_in[5];

    char* ws = (char*)d_ws;
    unsigned* Wpack = (unsigned*)(ws + OFF_WPACK);
    unsigned* Xpack = (unsigned*)(ws + OFF_XPACK);
    unsigned* Hpack = (unsigned*)(ws + OFF_HPACK);
    float*    gates = (float*)   (ws + OFF_GATES);
    float*    cbuf  = (float*)   (ws + OFF_C);
    float*    biasp = (float*)   (ws + OFF_BIAS);
    unsigned* barc  = (unsigned*)(ws + OFF_BAR);

    // deterministic re-init every call (graph-capture safe)
    hipMemsetAsync(Hpack, 0, SZ_HPACK, stream);          // h0 = 0 (bf16 zeros)
    hipMemsetAsync(cbuf,  0, SZ_C,     stream);          // c0 = 0
    hipMemsetAsync(barc,  0, 256,      stream);          // barrier counter
    hipMemsetAsync(d_out, 0, (size_t)out_size * sizeof(float), stream);

    pack_w_kernel<<<(G_ * (K_ / 2) + 255) / 256, 256, 0, stream>>>(Wih, Whh, Wpack);
    pack_x_kernel<<<(T_ * B_ * (I_ / 2) + 255) / 256, 256, 0, stream>>>(seq, Xpack);
    bias_kernel<<<(G_ + 255) / 256, 256, 0, stream>>>(bih, bhh, biasp);

    lstm_scan_kernel<<<GRID, 256, 0, stream>>>(Wpack, Xpack, Hpack, gates, cbuf,
                                               biasp, len, (float*)d_out, barc);
}
